// PointNetEncoder_49022756716560
// MI455X (gfx1250) — matrix operand order repaired and checked by reading, not compiled, not run on hardware
//
#include <hip/hip_runtime.h>
#include <hip/hip_bf16.h>
#include <math.h>

// ---------------------------------------------------------------------------
// Types for CDNA5 WMMA (wave32, v_wmma_f32_16x16x32_f16)
// ---------------------------------------------------------------------------
typedef __attribute__((ext_vector_type(16))) _Float16 v16h;
typedef __attribute__((ext_vector_type(8)))  _Float16 v8h;
typedef __attribute__((ext_vector_type(8)))  float    v8f;

#define BB    8
#define NN    2048
#define KNN   20
#define TT    49152            // 3 * B * N tokens for full-resolution layers
#define EPSV  1e-6f
#define BNEPS 1e-5f

// ===========================================================================
// KNN: for each (b,n) keep top-20 by dist = -||p_n - p_m||^2 (self included)
// ===========================================================================
__global__ void knn_kernel(const float* __restrict__ x, int* __restrict__ idx)
{
    __shared__ float bd[128 * KNN];
    __shared__ int   bi[128 * KNN];
    int tid = threadIdx.x;
    int gid = blockIdx.x * 128 + tid;          // over B*N (2048 % 128 == 0)
    int b = gid >> 11, n = gid & 2047;
    const float* xb = x + b * 3 * NN;
    float px = xb[n], py = xb[NN + n], pz = xb[2 * NN + n];
    float* d  = bd + tid * KNN;
    int*   ii = bi + tid * KNN;
#pragma unroll
    for (int j = 0; j < KNN; ++j) { d[j] = -3.0e38f; ii[j] = 0; }
    float thr = -3.0e38f; int slot = 0;
    for (int m = 0; m < NN; ++m) {
        float dx = px - xb[m], dy = py - xb[NN + m], dz = pz - xb[2 * NN + m];
        float dist = -(dx * dx + dy * dy + dz * dz);
        if (dist > thr) {
            d[slot] = dist; ii[slot] = m;
            thr = d[0]; slot = 0;
            for (int j = 1; j < KNN; ++j) if (d[j] < thr) { thr = d[j]; slot = j; }
        }
    }
    int* o = idx + (size_t)gid * KNN;
    for (int j = 0; j < KNN; ++j) o[j] = ii[j];
}

// ===========================================================================
// Fill helpers
// ===========================================================================
__global__ void fill_h(_Float16* p, size_t n)
{ size_t i = (size_t)blockIdx.x * blockDim.x + threadIdx.x; if (i < n) p[i] = (_Float16)0.f; }
__global__ void fill_f(float* p, size_t n)
{ size_t i = (size_t)blockIdx.x * blockDim.x + threadIdx.x; if (i < n) p[i] = 0.f; }

// ===========================================================================
// conv_pos: per (b,n,k) graph feature -> Wf projection norms -> BN statistics
// ===========================================================================
__global__ void cp_stats(const float* __restrict__ x, const int* __restrict__ idx,
                         const float* __restrict__ Wf,
                         float* __restrict__ S, float* __restrict__ S2)
{
    __shared__ float sW[63];
    __shared__ float ss[21], sq[21];
    int tid = threadIdx.x;
    if (tid < 63) sW[tid] = Wf[tid];
    if (tid < 21) { ss[tid] = 0.f; sq[tid] = 0.f; }
    __syncthreads();
    int gid = blockIdx.x * blockDim.x + tid;     // over B*N*K
    if (gid < BB * NN * KNN) {
        int k = gid % KNN; int bn = gid / KNN;
        int n = bn & 2047, b = bn >> 11;
        const float* xb = x + b * 3 * NN;
        int m = idx[(size_t)bn * KNN + k];
        float cx = xb[n], cy = xb[NN + n], cz = xb[2 * NN + n];
        float nx = xb[m], ny = xb[NN + m], nz = xb[2 * NN + m];
        float f1x = nx - cx, f1y = ny - cy, f1z = nz - cz;
        float f3x = ny * cz - nz * cy, f3y = nz * cx - nx * cz, f3z = nx * cy - ny * cx;
#pragma unroll
        for (int o = 0; o < 21; ++o) {
            float w0 = sW[3 * o], w1 = sW[3 * o + 1], w2 = sW[3 * o + 2];
            float qx = w0 * f1x + w1 * cx + w2 * f3x;
            float qy = w0 * f1y + w1 * cy + w2 * f3y;
            float qz = w0 * f1z + w1 * cz + w2 * f3z;
            float nn = sqrtf(qx * qx + qy * qy + qz * qz) + EPSV;
            atomicAdd(&ss[o], nn);
            atomicAdd(&sq[o], nn * nn);
        }
    }
    __syncthreads();
    if (tid < 21) { atomicAdd(&S[tid], ss[tid]); atomicAdd(&S2[tid], sq[tid]); }
}

// conv_pos LLR + mean over k, writes token-major f16 activations (CinP = 32)
__global__ void cp_apply(const float* __restrict__ x, const int* __restrict__ idx,
                         const float* __restrict__ Wf, const float* __restrict__ Wd,
                         const float* __restrict__ gamma, const float* __restrict__ beta,
                         const float* __restrict__ mu, const float* __restrict__ rstd,
                         _Float16* __restrict__ Xout)
{
    __shared__ float sWf[63], sWd[63], sG[21], sB[21], sM[21], sR[21];
    int tid = threadIdx.x;
    if (tid < 63) { sWf[tid] = Wf[tid]; sWd[tid] = Wd[tid]; }
    if (tid < 21) { sG[tid] = gamma[tid]; sB[tid] = beta[tid]; sM[tid] = mu[tid]; sR[tid] = rstd[tid]; }
    __syncthreads();
    int gid = blockIdx.x * blockDim.x + tid;     // over B*N
    int b = gid >> 11, n = gid & 2047;
    const float* xb = x + b * 3 * NN;
    float cx = xb[n], cy = xb[NN + n], cz = xb[2 * NN + n];
    float acc[63];
#pragma unroll
    for (int i = 0; i < 63; ++i) acc[i] = 0.f;
    const int* id = idx + (size_t)gid * KNN;
    for (int k = 0; k < KNN; ++k) {
        int m = id[k];
        float nx = xb[m], ny = xb[NN + m], nz = xb[2 * NN + m];
        float f1x = nx - cx, f1y = ny - cy, f1z = nz - cz;
        float f3x = ny * cz - nz * cy, f3y = nz * cx - nx * cz, f3z = nx * cy - ny * cx;
#pragma unroll
        for (int o = 0; o < 21; ++o) {
            float w0 = sWf[3 * o], w1 = sWf[3 * o + 1], w2 = sWf[3 * o + 2];
            float qx = w0 * f1x + w1 * cx + w2 * f3x;
            float qy = w0 * f1y + w1 * cy + w2 * f3y;
            float qz = w0 * f1z + w1 * cz + w2 * f3z;
            float nn = sqrtf(qx * qx + qy * qy + qz * qz) + EPSV;
            float s = ((nn - sM[o]) * sR[o] * sG[o] + sB[o]) / nn;
            qx *= s; qy *= s; qz *= s;
            float u0 = sWd[3 * o], u1 = sWd[3 * o + 1], u2 = sWd[3 * o + 2];
            float dx = u0 * f1x + u1 * cx + u2 * f3x;
            float dy = u0 * f1y + u1 * cy + u2 * f3y;
            float dz = u0 * f1z + u1 * cz + u2 * f3z;
            float dot = qx * dx + qy * dy + qz * dz;
            if (dot < 0.f) {
                float f = dot / (dx * dx + dy * dy + dz * dz + EPSV);
                qx -= f * dx; qy -= f * dy; qz -= f * dz;
            }
            acc[3 * o] += qx; acc[3 * o + 1] += qy; acc[3 * o + 2] += qz;
        }
    }
    const float inv = 1.0f / (float)KNN;
#pragma unroll
    for (int o = 0; o < 21; ++o)
#pragma unroll
        for (int v = 0; v < 3; ++v) {
            size_t t = (size_t)(b * 3 + v) * NN + n;
            Xout[t * 32 + o] = (_Float16)(acc[3 * o + v] * inv);
        }
}

// ===========================================================================
// Weight fp32 -> padded f16 (row-major Cout x CinP)
// ===========================================================================
__global__ void conv_w(const float* __restrict__ W, _Float16* __restrict__ Wh,
                       int Cout, int Cin, int CinP, int total)
{
    int gid = blockIdx.x * blockDim.x + threadIdx.x;
    if (gid >= total) return;
    int o = gid / CinP, c = gid - o * CinP;
    Wh[gid] = (o < Cout && c < Cin) ? (_Float16)W[o * Cin + c] : (_Float16)0.f;
}

// ===========================================================================
// Generic f16 WMMA GEMM:  Y[CoutP x Tp] (f32) = W[CoutP x CinP] * X^T
// X is token-major [Tp x CinP].  One wave per 16x16 tile, K-steps of 32.
// A fragment: lane row = lane&15, g = lane>>4; halves 0..7 = K[16g,16g+8),
//             halves 8..15 = K[16g+8,16g+16)  -> two 16B loads.
// B fragment: lanes 0..15 carry K 0..15, lanes 16..31 carry K 16..31,
//             contiguous per lane -> one 32B load from token-major X.
// ===========================================================================
__global__ void wmma_gemm(const _Float16* __restrict__ W, const _Float16* __restrict__ X,
                          float* __restrict__ Y, int CinP, int Tp, int CoutP)
{
    int lane   = threadIdx.x & 31;
    int wave   = threadIdx.x >> 5;
    int tilesT = Tp >> 4;
    int nTiles = tilesT * (CoutP >> 4);
    int tile   = blockIdx.x * (blockDim.x >> 5) + wave;
    if (tile >= nTiles) return;                 // wave-uniform: EXEC stays all-ones
    int to = tile / tilesT;
    int tt = tile - to * tilesT;
    int r  = lane & 15;
    int g  = lane >> 4;
    const _Float16* wrow  = W + (size_t)(to * 16 + r) * CinP + g * 16;
    const _Float16* wrow2 = wrow + 8;
    const _Float16* xrow  = X + (size_t)(tt * 16 + r) * CinP + g * 16;
    v8f acc = {};
    for (int k0 = 0; k0 < CinP; k0 += 32) {
        v8h  a0 = *(const v8h*)(wrow  + k0);
        v8h  a1 = *(const v8h*)(wrow2 + k0);
        v16h bb = *(const v16h*)(xrow + k0);
        __builtin_prefetch(xrow + k0 + 32, 0, 1);   // global_prefetch (speculative OOB ok)
        v16h aa;
#pragma unroll
        for (int i = 0; i < 8; ++i) { aa[i] = a0[i]; aa[i + 8] = a1[i]; }
        acc = __builtin_amdgcn_wmma_f32_16x16x32_f16(
            /*neg_a=*/false, aa, /*neg_b=*/false, bb,
            /*c_mod=*/(short)0, acc, /*reuse_a=*/false, /*reuse_b=*/false);
    }
    float* yo = Y + (size_t)(to * 16 + g * 8) * Tp + tt * 16 + r;   // D: VGPR j -> row j+8g
#pragma unroll
    for (int j = 0; j < 8; ++j) yo[(size_t)j * Tp] = acc[j];
}

// ===========================================================================
// BN statistics over vector norms: one block per channel
// token index t(b,v,m) = (b*3+v)*M + m
// ===========================================================================
__global__ void bn_stats(const float* __restrict__ Y, int Tp, int M, int BM,
                         float* __restrict__ S, float* __restrict__ S2)
{
    __shared__ float r1[256], r2[256];
    int o = blockIdx.x;
    const float* yr = Y + (size_t)o * Tp;
    float s = 0.f, sq = 0.f;
    for (int i = threadIdx.x; i < BM; i += 256) {
        int b = i / M, m = i - b * M;
        size_t t0 = (size_t)(b * 3) * M + m;
        float a0 = yr[t0], a1 = yr[t0 + M], a2 = yr[t0 + 2 * M];
        float nn = sqrtf(a0 * a0 + a1 * a1 + a2 * a2) + EPSV;
        s += nn; sq += nn * nn;
    }
    r1[threadIdx.x] = s; r2[threadIdx.x] = sq;
    __syncthreads();
    for (int w = 128; w > 0; w >>= 1) {
        if (threadIdx.x < w) { r1[threadIdx.x] += r1[threadIdx.x + w]; r2[threadIdx.x] += r2[threadIdx.x + w]; }
        __syncthreads();
    }
    if (threadIdx.x == 0) { S[o] = r1[0]; S2[o] = r2[0]; }
}

__global__ void bn_finalize(const float* __restrict__ S, const float* __restrict__ S2,
                            float cntInv, int C, float* __restrict__ mu, float* __restrict__ rstd)
{
    int i = blockIdx.x * blockDim.x + threadIdx.x;
    if (i >= C) return;
    float m = S[i] * cntInv;
    float v = S2[i] * cntInv - m * m;
    if (v < 0.f) v = 0.f;
    mu[i] = m; rstd[i] = rsqrtf(v + BNEPS);
}

// ===========================================================================
// LLR elementwise: BN-scale + (optional) direction projection.
// Writes token-major f16 next-layer input and/or keeps fp32 (may alias Yf).
// ===========================================================================
__global__ void llr_apply(const float* __restrict__ Yf, const float* __restrict__ Yd,
                          int Tp, int M, int Cout,
                          const float* __restrict__ mu, const float* __restrict__ rstd,
                          const float* __restrict__ gamma, const float* __restrict__ beta,
                          _Float16* __restrict__ Xout, int CinPn, int chanoff,
                          float* __restrict__ Ykeep)
{
    int gid = blockIdx.x * blockDim.x + threadIdx.x;
    int total = Cout * 8 * M;
    if (gid >= total) return;
    int o = gid / (8 * M); int r = gid - o * 8 * M;
    int b = r / M, m = r - b * M;
    size_t t0 = (size_t)(b * 3) * M + m;
    const float* yf = Yf + (size_t)o * Tp;
    float q0 = yf[t0], q1 = yf[t0 + M], q2 = yf[t0 + 2 * M];
    float nn = sqrtf(q0 * q0 + q1 * q1 + q2 * q2) + EPSV;
    float s = ((nn - mu[o]) * rstd[o] * gamma[o] + beta[o]) / nn;
    q0 *= s; q1 *= s; q2 *= s;
    if (Yd) {
        const float* yd = Yd + (size_t)o * Tp;
        float d0 = yd[t0], d1 = yd[t0 + M], d2 = yd[t0 + 2 * M];
        float dot = q0 * d0 + q1 * d1 + q2 * d2;
        if (dot < 0.f) {
            float f = dot / (d0 * d0 + d1 * d1 + d2 * d2 + EPSV);
            q0 -= f * d0; q1 -= f * d1; q2 -= f * d2;
        }
    }
    if (Ykeep) {
        float* yk = Ykeep + (size_t)o * Tp;
        yk[t0] = q0; yk[t0 + M] = q1; yk[t0 + 2 * M] = q2;
    }
    if (Xout) {
        Xout[t0 * CinPn + chanoff + o]           = (_Float16)q0;
        Xout[(t0 + M) * CinPn + chanoff + o]     = (_Float16)q1;
        Xout[(t0 + 2 * M) * CinPn + chanoff + o] = (_Float16)q2;
    }
}

// ===========================================================================
// Mean over m per (channel, b, v): one block per (o, bv)
// ===========================================================================
__global__ void pool_mean(const float* __restrict__ Y, int Tp, int M,
                          _Float16* __restrict__ Xout, int CinPn, int chanoff,
                          float* __restrict__ keep)
{
    __shared__ float rr[128];
    int blk = blockIdx.x;
    int o = blk / 24, bv = blk - o * 24;
    const float* yr = Y + (size_t)o * Tp + (size_t)bv * M;
    float s = 0.f;
    for (int i = threadIdx.x; i < M; i += 128) s += yr[i];
    rr[threadIdx.x] = s;
    __syncthreads();
    for (int w = 64; w > 0; w >>= 1) {
        if (threadIdx.x < w) rr[threadIdx.x] += rr[threadIdx.x + w];
        __syncthreads();
    }
    if (threadIdx.x == 0) {
        float mean = rr[0] / (float)M;
        if (Xout) Xout[(size_t)bv * CinPn + chanoff + o] = (_Float16)mean;
        if (keep) keep[o * 24 + bv] = mean;
    }
}

// Broadcast per-(o,b,v) mean across all m into token-major f16 channels
__global__ void bcast_mean(const float* __restrict__ keep, _Float16* __restrict__ X,
                           int CinPn, int chanoff, int Cout, int M)
{
    int gid = blockIdx.x * blockDim.x + threadIdx.x;
    int total = Cout * 24 * M;
    if (gid >= total) return;
    int o = gid / (24 * M); int r = gid - o * 24 * M;
    int bv = r / M, m = r - bv * M;
    X[((size_t)bv * M + m) * CinPn + chanoff + o] = (_Float16)keep[o * 24 + bv];
}

// ===========================================================================
// s_fc3: tiny 21x85 linear on pooled features (VALU)
// ===========================================================================
__global__ void fc3_kernel(const _Float16* __restrict__ X5, const float* __restrict__ W,
                           float* __restrict__ gv)
{
    int gid = blockIdx.x * blockDim.x + threadIdx.x;
    if (gid >= BB * 21 * 3) return;
    int b = gid / 63; int r = gid - b * 63;
    int o = r / 3, v = r - o * 3;
    const _Float16* xr = X5 + (size_t)(b * 3 + v) * 96;
    float s = 0.f;
    for (int c = 0; c < 85; ++c) s += W[o * 85 + c] * (float)xr[c];
    gv[(b * 21 + o) * 3 + v] = s;
}

// Concat for conv2 input: ch 0..20 = h2, 21..41 = broadcast g, 42..63 = 0
__global__ void concat2_kernel(const _Float16* __restrict__ X2, const float* __restrict__ gv,
                               _Float16* __restrict__ XO)
{
    int gid = blockIdx.x * blockDim.x + threadIdx.x;
    if (gid >= TT * 64) return;
    int t = gid >> 6, c = gid & 63;
    int b = t / (3 * NN), v = (t / NN) % 3;
    _Float16 val;
    if (c < 21)      val = X2[(size_t)t * 32 + c];
    else if (c < 42) val = (_Float16)gv[(b * 21 + (c - 21)) * 3 + v];
    else             val = (_Float16)0.f;
    XO[gid] = val;
}

// Copy z (rows 0..2 of Y layout) into d_out in (B,3,3,N) order
__global__ void zcopy_kernel(const float* __restrict__ Z, int Tp, float* __restrict__ out)
{
    int gid = blockIdx.x * blockDim.x + threadIdx.x;
    if (gid >= BB * 3 * 3 * NN) return;
    int b = gid / (9 * NN); int r = gid - b * 9 * NN;
    int ich = r / (3 * NN); int r2 = r - ich * 3 * NN;
    int v = r2 / NN, m = r2 - v * NN;
    out[gid] = Z[(size_t)ich * Tp + (size_t)(b * 3 + v) * NN + m];
}

// Final: hstd[b,i,k,m] = sum_j h6[b,i,j,m] * z[b,j,k,m]; out = max over m
__global__ void final_kernel(const _Float16* __restrict__ H6, const float* __restrict__ Z,
                             int Tp, float* __restrict__ out)
{
    __shared__ float red[3][64];
    int b = blockIdx.x / 682, i = blockIdx.x - b * 682;
    float mx0 = -3.0e38f, mx1 = -3.0e38f, mx2 = -3.0e38f;
    for (int m = threadIdx.x; m < NN; m += 64) {
        size_t tb = (size_t)(b * 3) * NN + m;
        float h0 = (float)H6[tb * 704 + i];
        float h1 = (float)H6[(tb + NN) * 704 + i];
        float h2 = (float)H6[(tb + 2 * NN) * 704 + i];
#pragma unroll
        for (int k = 0; k < 3; ++k) {
            float z0 = Z[0 * (size_t)Tp + (size_t)(b * 3 + k) * NN + m];
            float z1 = Z[1 * (size_t)Tp + (size_t)(b * 3 + k) * NN + m];
            float z2 = Z[2 * (size_t)Tp + (size_t)(b * 3 + k) * NN + m];
            float s = h0 * z0 + h1 * z1 + h2 * z2;
            if (k == 0) mx0 = fmaxf(mx0, s);
            else if (k == 1) mx1 = fmaxf(mx1, s);
            else mx2 = fmaxf(mx2, s);
        }
    }
    red[0][threadIdx.x] = mx0; red[1][threadIdx.x] = mx1; red[2][threadIdx.x] = mx2;
    __syncthreads();
    for (int w = 32; w > 0; w >>= 1) {
        if (threadIdx.x < w)
#pragma unroll
            for (int k = 0; k < 3; ++k)
                red[k][threadIdx.x] = fmaxf(red[k][threadIdx.x], red[k][threadIdx.x + w]);
        __syncthreads();
    }
    if (threadIdx.x == 0)
#pragma unroll
        for (int k = 0; k < 3; ++k)
            out[b * 2046 + i * 3 + k] = red[k][0];
}

// ===========================================================================
// Host-side generic LLR layer: convert weights, 2 WMMA GEMMs, BN stats,
// finalize, (zero next X), elementwise apply.
// ===========================================================================
struct Scratch {
    _Float16 *WA, *WB;
    float *YA, *YB, *S, *S2, *MU, *RS;
};

static void run_llr(hipStream_t st, const Scratch& sc,
                    const float* Wf, const float* Wd, const float* gamma, const float* beta,
                    int Cin, int Cout, int CinP, int CoutP,
                    const _Float16* Xin, int M,
                    _Float16* Xout, int CinPn, size_t zeroN,
                    float* Ykeep)
{
    int Tp = (M > 1) ? 24 * M : 32;
    int nw = CoutP * CinP;
    conv_w<<<(nw + 255) / 256, 256, 0, st>>>(Wf, sc.WA, Cout, Cin, CinP, nw);
    bool hasD = (Wd != nullptr);
    if (hasD) conv_w<<<(nw + 255) / 256, 256, 0, st>>>(Wd, sc.WB, Cout, Cin, CinP, nw);
    int tiles = (CoutP / 16) * (Tp / 16);
    int blocks = (tiles + 3) / 4;
    wmma_gemm<<<blocks, 128, 0, st>>>(sc.WA, Xin, sc.YA, CinP, Tp, CoutP);
    if (hasD) wmma_gemm<<<blocks, 128, 0, st>>>(sc.WB, Xin, sc.YB, CinP, Tp, CoutP);
    int BM = 8 * M;
    bn_stats<<<Cout, 256, 0, st>>>(sc.YA, Tp, M, BM, sc.S, sc.S2);
    bn_finalize<<<(Cout + 63) / 64, 64, 0, st>>>(sc.S, sc.S2, 1.0f / (float)BM, Cout, sc.MU, sc.RS);
    if (Xout && zeroN) fill_h<<<(unsigned)((zeroN + 255) / 256), 256, 0, st>>>(Xout, zeroN);
    int total = Cout * BM;
    llr_apply<<<(total + 255) / 256, 256, 0, st>>>(sc.YA, hasD ? sc.YB : nullptr, Tp, M, Cout,
                                                   sc.MU, sc.RS, gamma, beta,
                                                   Xout, CinPn, 0, Ykeep);
}

// ===========================================================================
extern "C" void kernel_launch(void* const* d_in, const int* in_sizes, int n_in,
                              void* d_out, int out_size, void* d_ws, size_t ws_size,
                              hipStream_t stream)
{
    (void)in_sizes; (void)n_in; (void)out_size; (void)ws_size;
    const float* X0 = (const float*)d_in[0];
#define P(i) ((const float*)d_in[i])

    // ---- carve workspace ----
    char* w = (char*)d_ws;
    size_t off = 0;
    auto alloc = [&](size_t bytes) -> char* {
        char* p = w + off; off += (bytes + 255) & ~(size_t)255; return p;
    };
    int*      IDX = (int*)alloc((size_t)BB * NN * KNN * 4);
    float*    S   = (float*)alloc(704 * 4);
    float*    S2  = (float*)alloc(704 * 4);
    float*    MU  = (float*)alloc(704 * 4);
    float*    RS  = (float*)alloc(704 * 4);
    float*    GV  = (float*)alloc(504 * 4);
    float*    HM  = (float*)alloc(341 * 24 * 4);
    _Float16* WA  = (_Float16*)alloc((size_t)352 * 704 * 2);
    _Float16* WB  = (_Float16*)alloc((size_t)352 * 704 * 2);
    _Float16* X2  = (_Float16*)alloc((size_t)TT * 32 * 2);    // persistent h2 (conv1 out)
    _Float16* XC  = (_Float16*)alloc((size_t)TT * 352 * 2);
    _Float16* XA  = (_Float16*)alloc((size_t)TT * 704 * 2);
    _Float16* XB  = (_Float16*)alloc((size_t)TT * 704 * 2);
    float*    YA  = (float*)alloc((size_t)352 * TT * 4);
    float*    YB  = (float*)alloc((size_t)352 * TT * 4);

    Scratch sc{WA, WB, YA, YB, S, S2, MU, RS};
    float* out = (float*)d_out;

    // ---- 1. KNN ----
    knn_kernel<<<BB * NN / 128, 128, 0, stream>>>(X0, IDX);

    // ---- 2. conv_pos (cin=3, VALU): stats -> finalize -> apply+mean_k -> XA ----
    fill_f<<<1, 32, 0, stream>>>(S, 21);
    fill_f<<<1, 32, 0, stream>>>(S2, 21);
    cp_stats<<<(BB * NN * KNN + 255) / 256, 256, 0, stream>>>(X0, IDX, P(1), S, S2);
    bn_finalize<<<1, 64, 0, stream>>>(S, S2, 1.0f / (float)(BB * NN * KNN), 21, MU, RS);
    fill_h<<<((size_t)TT * 32 + 255) / 256, 256, 0, stream>>>(XA, (size_t)TT * 32);
    cp_apply<<<BB * NN / 128, 128, 0, stream>>>(X0, IDX, P(1), P(2), P(3), P(4), MU, RS, XA);

    // ---- 3. conv1: 21->21, XA -> X2 (persistent h2) ----
    run_llr(stream, sc, P(5), P(6), P(7), P(8), 21, 21, 32, 32, XA, NN, X2, 32, (size_t)TT * 32, nullptr);
    // ---- 4. STN s1: 21->21, X2 -> XB ----
    run_llr(stream, sc, P(9), P(10), P(11), P(12), 21, 21, 32, 32, X2, NN, XB, 32, (size_t)TT * 32, nullptr);
    // ---- 5. s2: 21->42, XB -> XA ----
    run_llr(stream, sc, P(13), P(14), P(15), P(16), 21, 42, 32, 48, XB, NN, XA, 64, (size_t)TT * 64, nullptr);
    // ---- 6. s3: 42->341, XA -> YA (fp32, in-place keep) ----
    run_llr(stream, sc, P(17), P(18), P(19), P(20), 42, 341, 64, 352, XA, NN, nullptr, 0, 0, YA);
    // ---- 7. mean over N -> s4 input (32 x 352) in XB ----
    fill_h<<<(32 * 352 + 255) / 256, 256, 0, stream>>>(XB, (size_t)32 * 352);
    pool_mean<<<341 * 24, 128, 0, stream>>>(YA, TT, NN, XB, 352, 0, nullptr);
    // ---- 8. s4: 341->170 (M=1), XB -> XA ----
    run_llr(stream, sc, P(21), P(22), P(23), P(24), 341, 170, 352, 176, XB, 1, XA, 192, (size_t)32 * 192, nullptr);
    // ---- 9. s5: 170->85 (M=1), XA -> XB ----
    run_llr(stream, sc, P(25), P(26), P(27), P(28), 170, 85, 192, 96, XA, 1, XB, 96, (size_t)32 * 96, nullptr);
    // ---- 10. fc3: 85->21 -> GV ----
    fc3_kernel<<<2, 256, 0, stream>>>(XB, P(29), GV);
    // ---- 11. concat [h2, g] -> XA (T x 64) ----
    concat2_kernel<<<(TT * 64 + 255) / 256, 256, 0, stream>>>(X2, GV, XA);
    // ---- 12. conv2: 42->42, XA -> XB ----
    run_llr(stream, sc, P(30), P(31), P(32), P(33), 42, 42, 64, 48, XA, NN, XB, 64, (size_t)TT * 64, nullptr);
    // ---- 13. conv3 linear + vn_bn: 42->341, XB -> (h5 in YA, f16 in XB ch0..340 of 704) ----
    run_llr(stream, sc, P(34), nullptr, P(35), P(36), 42, 341, 64, 352, XB, NN, XB, 704, (size_t)TT * 704, YA);
    // ---- 14. channel mean of h5 -> broadcast into XB ch 341..681 ----
    pool_mean<<<341 * 24, 128, 0, stream>>>(YA, TT, NN, nullptr, 0, 0, HM);
    bcast_mean<<<(341 * 24 * NN + 255) / 256, 256, 0, stream>>>(HM, XB, 704, 341, 341, NN);
    // ---- 15. std_vn1: 682->341, XB -> XC ----
    run_llr(stream, sc, P(37), P(38), P(39), P(40), 682, 341, 704, 352, XB, NN, XC, 352, (size_t)TT * 352, nullptr);
    // ---- 16. std_vn2: 341->170, XC -> XA ----
    run_llr(stream, sc, P(41), P(42), P(43), P(44), 341, 170, 352, 176, XC, NN, XA, 192, (size_t)TT * 192, nullptr);
    // ---- 17. std_lin: 170->3 (plain WMMA GEMM), XA -> YA rows 0..2 ----
    conv_w<<<(16 * 192 + 255) / 256, 256, 0, stream>>>(P(45), WA, 3, 170, 192, 16 * 192);
    {
        int tiles = (16 / 16) * (TT / 16);
        wmma_gemm<<<(tiles + 3) / 4, 128, 0, stream>>>(WA, XA, YA, 192, TT, 16);
    }
    // ---- 18. z -> d_out[16368 ..) in (B,3,3,N) order ----
    zcopy_kernel<<<(BB * 9 * NN + 255) / 256, 256, 0, stream>>>(YA, TT, out + 16368);
    // ---- 19. hstd einsum + global max -> d_out[0 .. 16368) ----
    final_kernel<<<BB * 682, 64, 0, stream>>>(XB, YA, TT, out);
#undef P
}
